// ConvPoint_73177652789355
// MI455X (gfx1250) — matrix operand order repaired and checked by reading, not compiled, not run on hardware
//
#include <hip/hip_runtime.h>

// Problem constants (from reference)
#define BB   8
#define CC   64
#define OUTC 64
#define KS   16
#define NSZ  4096
#define KK   32

typedef __attribute__((ext_vector_type(16))) _Float16 v16h;
typedef __attribute__((ext_vector_type(8)))  _Float16 h8;
typedef __attribute__((ext_vector_type(8)))  float    v8f;

__device__ __forceinline__ v16h pack16(h8 lo, h8 hi) {
  v16h r;
#pragma unroll
  for (int i = 0; i < 8; ++i) { r[i] = lo[i]; r[i + 8] = hi[i]; }
  return r;
}

// Pre-convert weight (1024x64 f32, [k][o]) -> wT f16 [o][k] (B-fragment friendly).
__global__ __launch_bounds__(256) void prep_weight(const float* __restrict__ w,
                                                   _Float16* __restrict__ wT) {
  int idx = blockIdx.x * 256 + threadIdx.x;           // 65536 elements
  int o = idx >> 10, k = idx & 1023;
  wT[idx] = (_Float16)w[k * OUTC + o];
}

__global__ __launch_bounds__(256) void copy_support(const float* __restrict__ s,
                                                    float* __restrict__ dst, int n) {
  int i = blockIdx.x * 256 + threadIdx.x;
  if (i < n) dst[i] = s[i];
}

__global__ __launch_bounds__(256)
void convpoint_main(const float* __restrict__ input,     // (B,C,NS,K)
                    const float* __restrict__ points,    // (B,3,NS,K)
                    const float* __restrict__ support,   // (B,3,NS)
                    const _Float16* __restrict__ wT,     // (OUT,1024) f16
                    const float* __restrict__ bias,      // (OUT)
                    const float* __restrict__ centers,   // (3,KS)
                    const float* __restrict__ W1, const float* __restrict__ b1,
                    const float* __restrict__ W2, const float* __restrict__ b2,
                    const float* __restrict__ W3, const float* __restrict__ b3,
                    float* __restrict__ out)             // (B,OUT,NS)
{
  __shared__ float sS1[16 * 3];                          // folded layer-1 weights
  __shared__ float sc1[16];                              // folded layer-1 bias
  __shared__ float sb2[16], sb3[16];
  __shared__ float ssup[48], sbias[64];
  __shared__ __align__(16) _Float16 sW2B[16 * 40];       // W2^T, K zero-padded to 32
  __shared__ __align__(16) _Float16 sW3B[16 * 40];       // W3^T, K zero-padded to 32
  __shared__ __align__(16) float ptsv[3 * 16 * 32];      // [d][n][k]
  __shared__ float sumsq[16 * 32];
  __shared__ float smaxi[16];                            // 1/maxi
  __shared__ __align__(16) _Float16 matT[16 * 16 * 40];  // [n][m][k], row stride 40
  __shared__ __align__(16) unsigned char ubuf[512 * 40 * 2]; // hbuf(40KB) aliases feats(33KB)
  __shared__ __align__(16) float part[8 * 256];
  __shared__ float outl[64 * 17];                        // [o][n] padded

  _Float16* hbuf  = (_Float16*)ubuf;   // [p][40], p = n*32+k  (stage 3)
  _Float16* feats = (_Float16*)ubuf;   // [n][1032]            (stages 4-5)

  const int tid  = threadIdx.x;
  const int wave = tid >> 5;
  const int lane = tid & 31;
  const int hs   = lane >> 4;   // half-wave select
  const int l16  = lane & 15;

  const int b  = blockIdx.x >> 8;
  const int n0 = (blockIdx.x & 255) << 4;

  // Prefetch the streamed input tile (64c x 16n rows of 128B) into cache.
  for (int rr = tid; rr < CC * 16; rr += 256) {
    int c = rr >> 4, n = rr & 15;
    __builtin_prefetch(input + ((size_t)(b * CC + c) * NSZ + n0 + n) * KK, 0, 1);
  }

  // ---- stage 0: constants ----
  if (tid < 64) sbias[tid] = bias[tid];
  if (tid < 48)
    ssup[tid] = support[(size_t)(b * 3 + (tid >> 4)) * NSZ + n0 + (tid & 15)];
  if (tid < 16) {
    sb2[tid] = b2[tid];
    sb3[tid] = b3[tid];
    float s0 = 0.f, s1 = 0.f, s2 = 0.f, c = b1[tid];
#pragma unroll
    for (int m = 0; m < 16; ++m) {
      s0 += W1[tid * 48 + m];
      s1 += W1[tid * 48 + 16 + m];
      s2 += W1[tid * 48 + 32 + m];
    }
#pragma unroll
    for (int i = 0; i < 48; ++i) c -= W1[tid * 48 + i] * centers[i];
    sS1[tid * 3 + 0] = s0; sS1[tid * 3 + 1] = s1; sS1[tid * 3 + 2] = s2;
    sc1[tid] = c;
  }
  // W2^T / W3^T as zero-K-padded B-fragment layouts: sW?B[j][k] = W?[j][k], k<16, else 0
  {
    int i = tid, j = i >> 5, k = i & 31;
    if (i < 512) {
      _Float16 z = (_Float16)0.f;
      sW2B[j * 40 + k] = (k < 16) ? (_Float16)W2[j * 16 + k] : z;
      sW3B[j * 40 + k] = (k < 16) ? (_Float16)W3[j * 16 + k] : z;
    }
    i += 256; j = i >> 5; k = i & 31;
    _Float16 z = (_Float16)0.f;
    sW2B[j * 40 + k] = (k < 16) ? (_Float16)W2[j * 16 + k] : z;
    sW3B[j * 40 + k] = (k < 16) ? (_Float16)W3[j * 16 + k] : z;
  }
  __syncthreads();

  // ---- stage 1: relative coords ----
  for (int e = tid; e < 3 * 512; e += 256) {
    int d = e >> 9, r = e & 511, n = r >> 5, k = r & 31;
    float v = points[((size_t)(b * 3 + d) * NSZ + n0 + n) * KK + k] - ssup[d * 16 + n];
    ptsv[e] = v;
  }
  __syncthreads();

  // ---- stage 2: per-point max radius ----
  for (int p = tid; p < 512; p += 256) {
    float x = ptsv[p], y = ptsv[512 + p], z = ptsv[1024 + p];
    sumsq[p] = x * x + y * y + z * z;
  }
  __syncthreads();
  if (tid < 16) {
    float mx = 0.f;
#pragma unroll
    for (int k = 0; k < 32; ++k) mx = fmaxf(mx, sumsq[tid * 32 + k]);
    float mm = sqrtf(mx);
    if (mm == 0.f) mm = 1.f;
    smaxi[tid] = 1.f / mm;
  }
  __syncthreads();

  // ---- stage 3a: folded layer-1 on VALU -> hbuf rows (K zero-padded to 32) ----
  for (int p = tid; p < 512; p += 256) {
    int n = p >> 5;
    float inv = smaxi[n];
    float pd0 = ptsv[p] * inv, pd1 = ptsv[512 + p] * inv, pd2 = ptsv[1024 + p] * inv;
    h8 lo, hi, zz;
#pragma unroll
    for (int j = 0; j < 8; ++j) {
      float v0 = sc1[j] + pd0 * sS1[j * 3] + pd1 * sS1[j * 3 + 1] + pd2 * sS1[j * 3 + 2];
      float v1 = sc1[8 + j] + pd0 * sS1[(8 + j) * 3] + pd1 * sS1[(8 + j) * 3 + 1]
                            + pd2 * sS1[(8 + j) * 3 + 2];
      lo[j] = (_Float16)fmaxf(v0, 0.f);
      hi[j] = (_Float16)fmaxf(v1, 0.f);
      zz[j] = (_Float16)0.f;
    }
    *(h8*)&hbuf[p * 40]      = lo;
    *(h8*)&hbuf[p * 40 + 8]  = hi;
    *(h8*)&hbuf[p * 40 + 16] = zz;
    *(h8*)&hbuf[p * 40 + 24] = zz;
  }
  __syncthreads();

  // ---- stage 3b: layers 2 & 3 via WMMA; wave owns 4 row-tiles of 16 ----
  {
    const _Float16* b2p = &sW2B[l16 * 40 + hs * 8];
    v16h bf2 = pack16(*(const h8*)b2p, *(const h8*)(b2p + 16));
    const _Float16* b3p = &sW3B[l16 * 40 + hs * 8];
    v16h bf3 = pack16(*(const h8*)b3p, *(const h8*)(b3p + 16));
#pragma unroll
    for (int tt = 0; tt < 4; ++tt) {
      int t = wave * 4 + tt;
      const _Float16* ap = &hbuf[(t * 16 + l16) * 40 + hs * 8];
      v16h a = pack16(*(const h8*)ap, *(const h8*)(ap + 16));
      v8f acc = {};
      acc = __builtin_amdgcn_wmma_f32_16x16x32_f16(false, a, false, bf2,
                                                   (short)0, acc, false, false);
      // relu(acc + b2) -> h2 back into same rows (wave-private, DS in-order)
#pragma unroll
      for (int r = 0; r < 8; ++r) {
        int p = t * 16 + r + hs * 8;
        hbuf[p * 40 + l16] = (_Float16)fmaxf(acc[r] + sb2[l16], 0.f);
      }
      const _Float16* ap2 = &hbuf[(t * 16 + l16) * 40 + hs * 8];
      v16h a2 = pack16(*(const h8*)ap2, *(const h8*)(ap2 + 16));
      v8f acc2 = {};
      acc2 = __builtin_amdgcn_wmma_f32_16x16x32_f16(false, a2, false, bf3,
                                                    (short)0, acc2, false, false);
      // relu(acc2 + b3) -> matT[n][m][k]
#pragma unroll
      for (int r = 0; r < 8; ++r) {
        int p = t * 16 + r + hs * 8;
        int n = p >> 5, k = p & 31;
        matT[(n * 16 + l16) * 40 + k] = (_Float16)fmaxf(acc2[r] + sb3[l16], 0.f);
      }
    }
  }
  __syncthreads();

  // ---- stage 4: feats[n] = input(64x32) @ mat[n](32x16) via WMMA ----
#pragma unroll
  for (int nn = 0; nn < 2; ++nn) {
    int n = wave * 2 + nn;
    const _Float16* bp = &matT[(n * 16 + l16) * 40 + hs * 8];
    v16h bf = pack16(*(const h8*)bp, *(const h8*)(bp + 16));
#pragma unroll
    for (int ct = 0; ct < 4; ++ct) {
      int c = ct * 16 + l16;
      const float* rp = input + ((size_t)(b * CC + c) * NSZ + n0 + n) * KK + hs * 8;
      float4 f0 = *(const float4*)(rp);
      float4 f1 = *(const float4*)(rp + 4);
      float4 f2 = *(const float4*)(rp + 16);
      float4 f3 = *(const float4*)(rp + 20);
      v16h a;
      a[0]  = (_Float16)f0.x; a[1]  = (_Float16)f0.y; a[2]  = (_Float16)f0.z; a[3]  = (_Float16)f0.w;
      a[4]  = (_Float16)f1.x; a[5]  = (_Float16)f1.y; a[6]  = (_Float16)f1.z; a[7]  = (_Float16)f1.w;
      a[8]  = (_Float16)f2.x; a[9]  = (_Float16)f2.y; a[10] = (_Float16)f2.z; a[11] = (_Float16)f2.w;
      a[12] = (_Float16)f3.x; a[13] = (_Float16)f3.y; a[14] = (_Float16)f3.z; a[15] = (_Float16)f3.w;
      v8f acc = {};
      acc = __builtin_amdgcn_wmma_f32_16x16x32_f16(false, a, false, bf,
                                                   (short)0, acc, false, false);
#pragma unroll
      for (int r = 0; r < 8; ++r) {
        int cl = r + hs * 8;
        feats[n * 1032 + (ct * 16 + cl) * 16 + l16] = (_Float16)acc[r];
      }
    }
  }
  __syncthreads();

  // ---- stage 5: out(16x64) = feats(16x1024) @ weight(1024x64) via WMMA ----
  {
    const int otile = wave & 3;
    const int kh    = wave >> 2;
    v8f acc = {};
    for (int kt = 0; kt < 16; ++kt) {
      int kq = (kh * 16 + kt) * 32;
      const _Float16* ap = &feats[l16 * 1032 + kq + hs * 8];
      v16h a = pack16(*(const h8*)(ap), *(const h8*)(ap + 16));
      const _Float16* bp = wT + (size_t)(otile * 16 + l16) * 1024 + kq + hs * 8;
      v16h bf = pack16(*(const h8*)(bp), *(const h8*)(bp + 16));
      acc = __builtin_amdgcn_wmma_f32_16x16x32_f16(false, a, false, bf,
                                                   (short)0, acc, false, false);
    }
#pragma unroll
    for (int r = 0; r < 8; ++r) part[wave * 256 + r * 32 + lane] = acc[r];
  }
  __syncthreads();

  if (wave < 4) {
#pragma unroll
    for (int r = 0; r < 8; ++r) {
      float v = part[wave * 256 + r * 32 + lane] + part[(wave + 4) * 256 + r * 32 + lane];
      int o = wave * 16 + l16;
      int n = r + hs * 8;
      outl[o * 17 + n] = v * (1.f / (float)KK) + sbias[o];
    }
  }
  __syncthreads();

  // coalesced writeback: thread t -> o = t/4, 4 consecutive n
  {
    int o = tid >> 2, nq = (tid & 3) * 4;
    float4 v;
    v.x = outl[o * 17 + nq + 0];
    v.y = outl[o * 17 + nq + 1];
    v.z = outl[o * 17 + nq + 2];
    v.w = outl[o * 17 + nq + 3];
    *(float4*)(out + (size_t)(b * OUTC + o) * NSZ + n0 + nq) = v;
  }
}

extern "C" void kernel_launch(void* const* d_in, const int* in_sizes, int n_in,
                              void* d_out, int out_size, void* d_ws, size_t ws_size,
                              hipStream_t stream) {
  const float* input   = (const float*)d_in[0];
  const float* points  = (const float*)d_in[1];
  const float* support = (const float*)d_in[2];
  const float* weight  = (const float*)d_in[3];
  const float* bias    = (const float*)d_in[4];
  const float* centers = (const float*)d_in[5];
  const float* W1 = (const float*)d_in[6];
  const float* b1 = (const float*)d_in[7];
  const float* W2 = (const float*)d_in[8];
  const float* b2 = (const float*)d_in[9];
  const float* W3 = (const float*)d_in[10];
  const float* b3 = (const float*)d_in[11];
  float*    out = (float*)d_out;
  _Float16* wT  = (_Float16*)d_ws;   // 64*1024 halfs = 128 KB scratch

  // 1) weight -> f16 [o][k] layout for direct B-fragment loads (L2-resident)
  prep_weight<<<256, 256, 0, stream>>>(weight, wT);

  // 2) main fused kernel: one workgroup per (b, 16-point tile)
  convpoint_main<<<BB * (NSZ / 16), 256, 0, stream>>>(
      input, points, support, wT, bias, centers, W1, b1, W2, b2, W3, b3, out);

  // 3) second tuple output: support_points passthrough
  const int supn = BB * 3 * NSZ;                  // 98304
  copy_support<<<(supn + 255) / 256, 256, 0, stream>>>(
      support, out + (size_t)BB * OUTC * NSZ, supn);
}
